// PointNet_56942676410387
// MI455X (gfx1250) — compile-verified, matrix operand order checked by reading, not run on hardware
//
#include <hip/hip_runtime.h>
#include <hip/hip_bf16.h>

typedef __attribute__((ext_vector_type(16))) _Float16 v16h;
typedef __attribute__((ext_vector_type(8)))  _Float16 v8h;
typedef __attribute__((ext_vector_type(4)))  _Float16 v4h;
typedef __attribute__((ext_vector_type(8)))  float    v8f;

#define NPTS 1000000
#define NSEG 50000

// ---------------------------------------------------------------------------
// Fragment loader for V_WMMA_F32_16X16X32_F16 (ISA 7.12.2, 16-bit A 16x32):
//   lane l: row = (l & 15), hi = l>>4
//   v16h elements 0..7  = row[k0 + hi*8 + 0..7]
//   v16h elements 8..15 = row[k0 + 16 + hi*8 + 0..7]
// Same striping for B (lane -> N column of W^T stored [Nout][Kpad]).
// Two contiguous 16B loads per lane (ds_load_b128 / global_load_b128).
// ---------------------------------------------------------------------------
__device__ __forceinline__ v16h load_frag(const _Float16* __restrict__ p,
                                          int strideHalves, int row, int k0,
                                          int lane) {
  int hi = lane >> 4;
  const _Float16* r = p + (size_t)row * strideHalves + k0 + (hi << 3);
  v8h lo = *(const v8h*)(r);
  v8h hh = *(const v8h*)(r + 16);
  v16h o;
#pragma unroll
  for (int i = 0; i < 8; ++i) { o[i] = lo[i]; o[8 + i] = hh[i]; }
  return o;
}

__device__ __forceinline__ v8f wmma_step(v16h a, v16h b, v8f acc) {
  return __builtin_amdgcn_wmma_f32_16x16x32_f16(false, a, false, b, (short)0,
                                                acc, false, false);
}

// One fused MLP layer: Hout = relu(Hin @ W + b).  Hin/Hout LDS f16, W^T global
// f16 [Nout][KSTEPS*32].  Wave w owns M-tile w; A-fragments for the whole K
// dimension are preloaded into registers and reused across all N-tiles, so the
// steady-state inner loop is 2x b128 (B) + 1x wmma.
template <int KSTEPS, int NTILES>
__device__ __forceinline__ void mlp_layer(const _Float16* Hin,
                                          const _Float16* __restrict__ Wt,
                                          const float* __restrict__ bias,
                                          int NoutReal, _Float16* Hout,
                                          int Nout, int wave, int lane) {
  const int Kpad = KSTEPS * 32;
  const int mt = wave, n = lane & 15, hi = lane >> 4;
  v16h a[KSTEPS];
#pragma unroll
  for (int ks = 0; ks < KSTEPS; ++ks)
    a[ks] = load_frag(Hin, Kpad, (mt << 4) + (lane & 15), ks << 5, lane);
#pragma unroll 1
  for (int nt = 0; nt < NTILES; ++nt) {
    const int nb = nt << 4;
    v8f acc = {};  // SRC2 starts at 0; bias folded into epilogue
#pragma unroll
    for (int ks = 0; ks < KSTEPS; ++ks) {
      v16h b = load_frag(Wt, Kpad, nb + (lane & 15), ks << 5, lane);
      acc = wmma_step(a[ks], b, acc);
    }
    float bv = (nb + n < NoutReal) ? bias[nb + n] : 0.0f;
    // D layout: VGPR v, lanes 0-15 -> M=v, lanes 16-31 -> M=v+8; N = lane&15
#pragma unroll
    for (int v = 0; v < 8; ++v) {
      float x = acc[v] + bv;
      x = x > 0.0f ? x : 0.0f;
      Hout[(size_t)((mt << 4) + v + (hi << 3)) * Nout + nb + n] = (_Float16)x;
    }
  }
}

// ---------------------------------------------------------------------------
// Weight pre-pass: fp32 [K][N] -> f16 transposed+padded [Npad][Kpad]
// ---------------------------------------------------------------------------
__global__ void convert_weight(const float* __restrict__ W,
                               _Float16* __restrict__ Wt,
                               int K, int N, int Kpad, int Npad) {
  int e = blockIdx.x * blockDim.x + threadIdx.x;
  if (e >= Kpad * Npad) return;
  int n = e / Kpad, k = e % Kpad;
  float v = (k < K && n < N) ? W[(size_t)k * N + n] : 0.0f;
  Wt[e] = (_Float16)v;
}

// ---------------------------------------------------------------------------
// Kernel 1: fused point MLP (11->64->128->256) + segment-max via atomicMax on
// non-negative IEEE bits.  128 points / block, 8 waves.
// ---------------------------------------------------------------------------
__global__ void __launch_bounds__(256) point_mlp_kernel(
    const float* __restrict__ x, const int* __restrict__ idx,
    const _Float16* __restrict__ W0t, const float* __restrict__ b0,
    const _Float16* __restrict__ W1t, const float* __restrict__ b1,
    const _Float16* __restrict__ W2t, const float* __restrict__ b2,
    float* __restrict__ g) {
  extern __shared__ _Float16 smem1[];
  _Float16* X  = smem1;            // [128][32]   4096 halves
  _Float16* H0 = X  + 128 * 32;    // [128][64]   8192 halves
  _Float16* H1 = H0 + 128 * 64;    // [128][128] 16384 halves
  int* sIdx = (int*)(H1 + 128 * 128);

  const int tid  = threadIdx.x;
  const int wave = tid >> 5;
  const int lane = tid & 31;
  const int p0   = blockIdx.x * 128;

  // Stage x tile, K padded 11->32, invalid rows zeroed.
  for (int e = tid; e < 128 * 32; e += 256) {
    int r = e >> 5, c = e & 31;
    int grow = p0 + r;
    float v = (c < 11 && grow < NPTS) ? x[(size_t)grow * 11 + c] : 0.0f;
    X[e] = (_Float16)v;
  }
  if (tid < 128) {
    int grow = p0 + tid;
    sIdx[tid] = (grow < NPTS) ? idx[grow] : 0;
  }
  __syncthreads();

  mlp_layer<1, 4>(X,  W0t, b0,  64, H0,  64, wave, lane);
  __syncthreads();
  mlp_layer<2, 8>(H0, W1t, b1, 128, H1, 128, wave, lane);
  __syncthreads();

  // Layer 2 (K=128, Nout=256): relu + atomic segment-max straight from regs.
  {
    const int mt = wave, n = lane & 15, hi = lane >> 4;
    v16h a[4];
#pragma unroll
    for (int ks = 0; ks < 4; ++ks)
      a[ks] = load_frag(H1, 128, (mt << 4) + (lane & 15), ks << 5, lane);
#pragma unroll 1
    for (int nt = 0; nt < 16; ++nt) {
      const int nb = nt << 4;
      v8f acc = {};
#pragma unroll
      for (int ks = 0; ks < 4; ++ks) {
        v16h b = load_frag(W2t, 128, nb + (lane & 15), ks << 5, lane);
        acc = wmma_step(a[ks], b, acc);
      }
      float bv = b2[nb + n];
#pragma unroll
      for (int v = 0; v < 8; ++v) {
        int rl = (mt << 4) + v + (hi << 3);
        int grow = p0 + rl;
        if (grow < NPTS) {
          float xv = acc[v] + bv;
          xv = xv > 0.0f ? xv : 0.0f;  // relu => uint bits monotonic
          int seg = sIdx[rl];
          atomicMax((unsigned int*)(g + (size_t)seg * 256 + nb + n),
                    __float_as_uint(xv));
        }
      }
    }
  }
}

// ---------------------------------------------------------------------------
// Kernel 2: fused segment MLP (256->256->128->13).  128 segments / block.
// ---------------------------------------------------------------------------
__global__ void __launch_bounds__(256) seg_mlp_kernel(
    const float* __restrict__ g,
    const _Float16* __restrict__ W3t, const float* __restrict__ b3,
    const _Float16* __restrict__ W4t, const float* __restrict__ b4,
    const _Float16* __restrict__ Wht, const float* __restrict__ bh,
    float* __restrict__ out) {
  extern __shared__ _Float16 smem2[];
  _Float16* G  = smem2;            // [128][256] 64KB
  _Float16* H3 = G + 128 * 256;    // [128][256] 64KB
  _Float16* H4 = G;                // [128][128] aliases G (dead after L3)

  const int tid  = threadIdx.x;
  const int wave = tid >> 5;
  const int lane = tid & 31;
  const int s0   = blockIdx.x * 128;

  // Stage g tile (f32 -> f16), vectorized: float4 in, 4x f16 out.
  for (int e = tid; e < 128 * 64; e += 256) {
    int r = e >> 6, c = (e & 63) << 2;
    int grow = s0 + r;
    float4 v = make_float4(0.0f, 0.0f, 0.0f, 0.0f);
    if (grow < NSEG) v = *(const float4*)(g + (size_t)grow * 256 + c);
    v4h h;
    h[0] = (_Float16)v.x; h[1] = (_Float16)v.y;
    h[2] = (_Float16)v.z; h[3] = (_Float16)v.w;
    *(v4h*)(G + r * 256 + c) = h;
  }
  __syncthreads();

  mlp_layer<8, 16>(G,  W3t, b3, 256, H3, 256, wave, lane);
  __syncthreads();
  mlp_layer<8, 8>(H3, W4t, b4, 128, H4, 128, wave, lane);
  __syncthreads();

  // Head: K=128, Nout padded 13->16, no relu, f32 store.
  {
    const int mt = wave, n = lane & 15, hi = lane >> 4;
    v16h a[4];
#pragma unroll
    for (int ks = 0; ks < 4; ++ks)
      a[ks] = load_frag(H4, 128, (mt << 4) + (lane & 15), ks << 5, lane);
    v8f acc = {};
#pragma unroll
    for (int ks = 0; ks < 4; ++ks) {
      v16h b = load_frag(Wht, 128, n, ks << 5, lane);
      acc = wmma_step(a[ks], b, acc);
    }
    float bv = (n < 13) ? bh[n] : 0.0f;
    if (n < 13) {
#pragma unroll
      for (int v = 0; v < 8; ++v) {
        int row = s0 + (mt << 4) + v + (hi << 3);
        if (row < NSEG) out[(size_t)row * 13 + n] = acc[v] + bv;
      }
    }
  }
}

// ---------------------------------------------------------------------------
extern "C" void kernel_launch(void* const* d_in, const int* in_sizes, int n_in,
                              void* d_out, int out_size, void* d_ws,
                              size_t ws_size, hipStream_t stream) {
  const float* x   = (const float*)d_in[0];
  const int*   idx = (const int*)d_in[1];
  const float* W0  = (const float*)d_in[2];  const float* b0 = (const float*)d_in[3];
  const float* W1  = (const float*)d_in[4];  const float* b1 = (const float*)d_in[5];
  const float* W2  = (const float*)d_in[6];  const float* b2 = (const float*)d_in[7];
  const float* W3  = (const float*)d_in[8];  const float* b3 = (const float*)d_in[9];
  const float* W4  = (const float*)d_in[10]; const float* b4 = (const float*)d_in[11];
  const float* Wh  = (const float*)d_in[12]; const float* bh = (const float*)d_in[13];
  float* out = (float*)d_out;

  // Workspace carve-up (g pooled buffer + f16 transposed weights).
  char* ws = (char*)d_ws;
  float* g = (float*)ws;
  size_t off = (size_t)NSEG * 256 * sizeof(float);   // 51.2 MB, 256B aligned
  _Float16* W0t = (_Float16*)(ws + off); off += (size_t)64  * 32  * 2;
  _Float16* W1t = (_Float16*)(ws + off); off += (size_t)128 * 64  * 2;
  _Float16* W2t = (_Float16*)(ws + off); off += (size_t)256 * 128 * 2;
  _Float16* W3t = (_Float16*)(ws + off); off += (size_t)256 * 256 * 2;
  _Float16* W4t = (_Float16*)(ws + off); off += (size_t)128 * 256 * 2;
  _Float16* Wht = (_Float16*)(ws + off); off += (size_t)16  * 128 * 2;

  // g must be zero every call (atomicMax accumulator; empty segments -> 0).
  hipMemsetAsync(g, 0, (size_t)NSEG * 256 * sizeof(float), stream);

  // Weight pre-pass: transpose + pad + fp32->f16.
  auto cvt = [&](const float* W, _Float16* Wt, int K, int N, int Kp, int Np) {
    int n = Kp * Np;
    convert_weight<<<(n + 255) / 256, 256, 0, stream>>>(W, Wt, K, N, Kp, Np);
  };
  cvt(W0, W0t, 11,  64,  32,  64);
  cvt(W1, W1t, 64,  128, 64,  128);
  cvt(W2, W2t, 128, 256, 128, 256);
  cvt(W3, W3t, 256, 256, 256, 256);
  cvt(W4, W4t, 256, 128, 256, 128);
  cvt(Wh, Wht, 128, 13,  128, 16);

  // Fused point MLP + segment max.
  {
    size_t lds = (size_t)(128 * 32 + 128 * 64 + 128 * 128) * 2 + 128 * 4;
    int blocks = (NPTS + 127) / 128;  // 7813
    point_mlp_kernel<<<blocks, 256, lds, stream>>>(x, idx, W0t, b0, W1t, b1,
                                                   W2t, b2, g);
  }
  // Fused segment MLP + head.
  {
    size_t lds = (size_t)(128 * 256 + 128 * 256) * 2;  // 128 KB
    int blocks = (NSEG + 127) / 128;  // 391
    seg_mlp_kernel<<<blocks, 256, lds, stream>>>(g, W3t, b3, W4t, b4, Wht, bh,
                                                 out);
  }
}
